// HanMeasNoteDecoder_77764677862237
// MI455X (gfx1250) — compile-verified, matrix operand order checked by reading, not run on hardware
//
#include <hip/hip_runtime.h>
#include <hip/hip_bf16.h>

typedef __attribute__((ext_vector_type(16))) __bf16 v16bf;
typedef __attribute__((ext_vector_type(8)))  __bf16 v8bf;
typedef __attribute__((ext_vector_type(8)))  float  v8f;

#define Bx   8
#define Nn   1024
#define NBt  128
#define NMs  32
#define OLW  717
#define BTW  461

// ---- workspace float offsets (f32 buffers) ----
#define WS_PZ      0u
#define WS_APZ_OL  512u
#define WS_APZ_BT  4608u
#define WS_ANOTE   8704u
#define WS_ABE_OL  4203008u
#define WS_ABE_BT  4727296u
#define WS_MAUG    5251584u
#define WS_AME_OL  5284864u
#define WS_AME_BT  5415936u
#define WS_OUT     5547008u
#define WS_SIM     5637120u
// ---- bf16 staging buffers (offsets in float units) ----
#define WS_NOTE16   5645312u   // 8192x256 bf16
#define WS_BEAT16   6693888u   // 1024x256 bf16
#define WS_MAUG16   6824960u   // 256x160 bf16
#define WS_WOLN16   6845440u   // 512x256 bf16 (ol_Wih cols 0:256)
#define WS_WOLB16   6910976u   // 512x256 bf16 (ol_Wih cols 256:512)
#define WS_WOLM16   6976512u   // 512x160 bf16 (ol_Wih cols 512:642 padded)
#define WS_WBTB16   7017472u   // 512x256 bf16 (bt_Wih cols 0:256)
#define WS_WBTM16   7058432u   // 512x160 bf16 (bt_Wih cols 256:386 padded)
#define WS_WOLHH16  7099392u   // 512x128 bf16
#define WS_WBTHH16  7132160u   // 512x128 bf16
#define WS_WPBT16   7164928u   // 512x12 bf16 (bt_Wih cols 386:397 padded)

__device__ __forceinline__ float sigmf(float x) { return 1.f / (1.f + __expf(-x)); }
__device__ __forceinline__ v8f vzero8() { v8f v;
#pragma unroll
  for (int r = 0; r < 8; ++r) v[r] = 0.f; return v; }

// Build a 16x32 fragment from contiguous bf16 memory:
// a[0..7] = base[k0..k0+7], a[8..15] = base[k0+16..k0+23]
__device__ __forceinline__ v16bf ldfrag(const __bf16* base, int k0) {
  v8bf lo = *(const v8bf*)(base + k0);
  v8bf hi = *(const v8bf*)(base + k0 + 16);
  return __builtin_shufflevector(lo, hi, 0, 1, 2, 3, 4, 5, 6, 7,
                                 8, 9, 10, 11, 12, 13, 14, 15);
}

// ---------------------------------------------------------------------------
// f32 -> bf16 staging with K padding
// ---------------------------------------------------------------------------
__global__ __launch_bounds__(256)
void k_cvt(const float* __restrict__ src, int lds, int C, int Kp, int R,
           __bf16* __restrict__ dst) {
  int total = R * Kp;
  for (int idx = blockIdx.x * 256 + threadIdx.x; idx < total;
       idx += gridDim.x * 256) {
    int r = idx / Kp, c = idx % Kp;
    dst[idx] = (c < C) ? (__bf16)src[(size_t)r * lds + c] : (__bf16)0.f;
  }
}

// ---------------------------------------------------------------------------
// bf16 WMMA GEMM: C[M,512] = X[M,Kp] @ W[512,Kp].T (+ per-batch bias row).
// bias != null: C[row][col] += bias[(row/rowsPerB)*512 + col]
// Wave computes 16 rows x 64 cols (4 tiles); guard-free b128 inner loop.
// ---------------------------------------------------------------------------
__global__ __launch_bounds__(256)
void k_gemm16(const __bf16* __restrict__ X, int ldx,
              const __bf16* __restrict__ W, int ldw,
              float* __restrict__ C, int M, int Ksteps,
              const float* __restrict__ bias, int rowsPerB) {
  int wave = blockIdx.x * 8 + (threadIdx.x >> 5);
  int lane = threadIdx.x & 31;
  int mt = wave >> 3, ng = wave & 7;           // 8 n-groups of 64 cols
  if (mt * 16 >= M) return;
  int m = lane & 15, hi = lane >> 4;
  const __bf16* xrow = X + (size_t)(mt * 16 + m) * ldx;
  const __bf16* wrow0 = W + (size_t)((ng * 4 + 0) * 16 + m) * ldw;
  const __bf16* wrow1 = W + (size_t)((ng * 4 + 1) * 16 + m) * ldw;
  const __bf16* wrow2 = W + (size_t)((ng * 4 + 2) * 16 + m) * ldw;
  const __bf16* wrow3 = W + (size_t)((ng * 4 + 3) * 16 + m) * ldw;
  v8f acc0 = vzero8(), acc1 = vzero8(), acc2 = vzero8(), acc3 = vzero8();
#pragma unroll 2
  for (int ks = 0; ks < Ksteps; ++ks) {
    int k0 = ks * 32 + hi * 8;
    v16bf a = ldfrag(xrow, k0);
    v16bf b0 = ldfrag(wrow0, k0);
    v16bf b1 = ldfrag(wrow1, k0);
    v16bf b2 = ldfrag(wrow2, k0);
    v16bf b3 = ldfrag(wrow3, k0);
    acc0 = __builtin_amdgcn_wmma_f32_16x16x32_bf16(false, a, false, b0, (short)0, acc0, false, false);
    acc1 = __builtin_amdgcn_wmma_f32_16x16x32_bf16(false, a, false, b1, (short)0, acc1, false, false);
    acc2 = __builtin_amdgcn_wmma_f32_16x16x32_bf16(false, a, false, b2, (short)0, acc2, false, false);
    acc3 = __builtin_amdgcn_wmma_f32_16x16x32_bf16(false, a, false, b3, (short)0, acc3, false, false);
  }
  float bv0 = 0.f, bv1 = 0.f, bv2 = 0.f, bv3 = 0.f;
  if (bias) {
    // 16-row tiles never straddle a batch (rowsPerB multiple of 16)
    const float* brow = bias + (size_t)((mt * 16) / rowsPerB) * 512 + ng * 64 + m;
    bv0 = brow[0]; bv1 = brow[16]; bv2 = brow[32]; bv3 = brow[48];
  }
#pragma unroll
  for (int r = 0; r < 8; ++r) {
    int row = mt * 16 + hi * 8 + r;
    float* crow = C + (size_t)row * 512 + ng * 64 + m;
    crow[0]  = acc0[r] + bv0;
    crow[16] = acc1[r] + bv1;
    crow[32] = acc2[r] + bv2;
    crow[48] = acc3[r] + bv3;
  }
}

// ---------------------------------------------------------------------------
// perform_z and pz/bias folding
// ---------------------------------------------------------------------------
__global__ __launch_bounds__(256)
void k_pz(const float* __restrict__ perf, const float* __restrict__ style_W,
          const float* __restrict__ style_b,
          const float* __restrict__ ol_Wih, const float* __restrict__ ol_bih,
          const float* __restrict__ ol_bhh,
          const float* __restrict__ bt_Wih, const float* __restrict__ bt_bih,
          const float* __restrict__ bt_bhh, float* __restrict__ ws) {
  __shared__ float pz[8][64];
  int tid = threadIdx.x;
  for (int idx = tid; idx < 512; idx += 256) {
    int b = idx >> 6, e = idx & 63;
    float s = style_b[e];
    for (int z = 0; z < 16; ++z) s += perf[b * 16 + z] * style_W[e * 16 + z];
    pz[b][e] = s;
    ws[WS_PZ + idx] = s;
  }
  __syncthreads();
  for (int idx = tid; idx < 8 * 512; idx += 256) {
    int b = idx >> 9, j = idx & 511;
    float so = ol_bih[j] + ol_bhh[j];
    float sb = bt_bih[j] + bt_bhh[j];
    for (int e = 0; e < 64; ++e) {
      float p = pz[b][e];
      so += p * ol_Wih[j * OLW + 653 + e];
      sb += p * bt_Wih[j * BTW + 397 + e];
    }
    ws[WS_APZ_OL + idx] = so;
    ws[WS_APZ_BT + idx] = sb;
  }
}

// ---------------------------------------------------------------------------
// Measure-level LSTM (32 sequential steps)
// ---------------------------------------------------------------------------
__global__ __launch_bounds__(256)
void k_measure(const float* __restrict__ measure_emb,
               const float* __restrict__ res_info,
               const float* __restrict__ md_Wih, const float* __restrict__ md_Whh,
               const float* __restrict__ md_bih, const float* __restrict__ md_bhh,
               const float* __restrict__ md_fcW, const float* __restrict__ md_fcb,
               float* __restrict__ ws, float* __restrict__ out) {
  __shared__ float x[8][202];
  __shared__ float h[8][64], c[8][64];
  __shared__ float z[8][256];
  __shared__ float prev2[8][2];
  __shared__ int zero_m[8];
  int tid = threadIdx.x;
  const float* pz = ws + WS_PZ;
  float* maug = ws + WS_MAUG;
  for (int idx = tid; idx < 8 * 64; idx += 256) {
    int b = idx >> 6, e = idx & 63;
    h[b][e] = 0.f; c[b][e] = 0.f;
    x[b][128 + e] = pz[b * 64 + e];
  }
  if (tid < 16) prev2[tid >> 1][tid & 1] = 0.f;
  __syncthreads();
#pragma unroll 1
  for (int m = 0; m < NMs; ++m) {
    for (int idx = tid; idx < 8 * 128; idx += 256) {
      int b = idx >> 7, e = idx & 127;
      float v = measure_emb[(b * NMs + m) * 128 + e];
      x[b][e] = v;
      maug[(b * NMs + m) * 130 + e] = v;
    }
    if (tid < 64) { int b = tid >> 3, e = tid & 7;
      x[b][192 + e] = res_info[(b * NMs + m) * 8 + e]; }
    if (tid < 16) { int b = tid >> 1, e = tid & 1; x[b][200 + e] = prev2[b][e]; }
    if (tid < 8)  { int b = tid; int zf = 1;
      for (int e = 0; e < 128; ++e)
        if (measure_emb[(b * NMs + m) * 128 + e] != 0.f) { zf = 0; break; }
      zero_m[b] = zf; }
    __syncthreads();
    for (int idx = tid; idx < 8 * 256; idx += 256) {
      int b = idx >> 8, j = idx & 255;
      float s = md_bih[j] + md_bhh[j];
      for (int k = 0; k < 202; ++k) s += md_Wih[j * 202 + k] * x[b][k];
      for (int k = 0; k < 64;  ++k) s += md_Whh[j * 64 + k] * h[b][k];
      z[b][j] = s;
    }
    __syncthreads();
    for (int idx = tid; idx < 8 * 64; idx += 256) {
      int b = idx >> 6, k = idx & 63;
      float c2 = sigmf(z[b][64 + k]) * c[b][k] + sigmf(z[b][k]) * tanhf(z[b][128 + k]);
      float h2 = sigmf(z[b][192 + k]) * tanhf(c2);
      c[b][k] = c2; h[b][k] = h2;
    }
    __syncthreads();
    if (tid < 16) {
      int b = tid >> 1, d = tid & 1;
      float s = md_fcb[d];
      for (int k = 0; k < 64; ++k) s += md_fcW[d * 64 + k] * h[b][k];
      prev2[b][d] = s;
      float msv = zero_m[b] ? 0.f : s;
      out[(size_t)Bx * Nn * 11 + (b * NMs + m) * 2 + d] = msv;
      maug[(b * NMs + m) * 130 + 128 + d] = msv;
    }
    __syncthreads();
  }
}

// ---------------------------------------------------------------------------
// Main sequential loop: persistent 256-thread workgroup, 1024 steps.
// ---------------------------------------------------------------------------
__global__ __launch_bounds__(256)
void k_mainloop(const int* __restrict__ beat_nums, const int* __restrict__ meas_nums,
                const float* __restrict__ ol_Wih,
                const float* __restrict__ fc_W,  const float* __restrict__ fc_b,
                const float* __restrict__ btf_W, const float* __restrict__ btf_b,
                const float* __restrict__ att_W, const float* __restrict__ att_b,
                const float* __restrict__ att_ctx, float* __restrict__ ws) {
  __shared__ float sh_h[8][128], sh_c[8][128], sh_th[8][128], sh_tc[8][128];
  __shared__ __bf16 sh_hbf[16][128];   // rows 8..15 stay zero
  __shared__ __bf16 sh_thbf[16][128];
  __shared__ float sh_z[8][512];
  __shared__ __bf16 Wp_ol[512][12];
  __shared__ float sh_prev[8][12];
  __shared__ float sh_cbr[8][10];
  __shared__ float sh_res[8][10];
  __shared__ float sh_red[8][32];
  __shared__ float sh_mx[8], sh_se[8], sh_tempo[8];
  __shared__ int sh_ch[8], sh_cb[8], sh_cm[8], sh_pb[8];
  __shared__ int sh_any;

  int tid = threadIdx.x, lane = tid & 31, wave = tid >> 5;
  int m = lane & 15, hi = lane >> 4;

  const float* Apz_bt = ws + WS_APZ_BT;      // still used if sh_cb row lacks bias? (folded into Abe_bt)
  (void)Apz_bt;
  const float* Anote  = ws + WS_ANOTE;       // includes ol biases + pz term
  const float* Abe_ol = ws + WS_ABE_OL;
  const float* Abe_bt = ws + WS_ABE_BT;      // includes bt biases + pz term
  const float* Ame_ol = ws + WS_AME_OL;
  const float* Ame_bt = ws + WS_AME_BT;
  const __bf16* wolhh16 = (const __bf16*)(ws + WS_WOLHH16);
  const __bf16* wbthh16 = (const __bf16*)(ws + WS_WBTHH16);
  const __bf16* wpbt16  = (const __bf16*)(ws + WS_WPBT16);
  float* outw = ws + WS_OUT;
  float* simw = ws + WS_SIM;

  for (int idx = tid; idx < 8 * 128; idx += 256) {
    int b = idx >> 7, k = idx & 127;
    sh_h[b][k] = 0.f; sh_c[b][k] = 0.f; sh_th[b][k] = 0.f; sh_tc[b][k] = 0.f;
  }
  for (int idx = tid; idx < 16 * 128; idx += 256) {
    sh_hbf[idx >> 7][idx & 127] = (__bf16)0.f;
    sh_thbf[idx >> 7][idx & 127] = (__bf16)0.f;
  }
  if (tid < 96)  sh_prev[tid / 12][tid % 12] = 0.f;
  if (tid < 80)  sh_cbr[tid / 10][tid % 10] = 0.f;
  for (int idx = tid; idx < 512 * 11; idx += 256) {
    int j = idx / 11, p = idx % 11;
    Wp_ol[j][p] = (__bf16)ol_Wih[j * OLW + 642 + p];
  }
  __syncthreads();

  // persistent ol_Whh B fragments
  v16bf Bol[4][4];
#pragma unroll
  for (int t = 0; t < 4; ++t) {
    const __bf16* wrow = wolhh16 + (size_t)((wave * 4 + t) * 16 + m) * 128;
#pragma unroll
    for (int ks = 0; ks < 4; ++ks)
      Bol[t][ks] = ldfrag(wrow, ks * 32 + hi * 8);
  }

#pragma unroll 1
  for (int i = 0; i < Nn; ++i) {
    if (tid < 8) {
      int b = tid;
      int bn  = beat_nums[b * Nn + i];
      int bn0 = beat_nums[b * Nn];
      int bnp = (i > 0) ? beat_nums[b * Nn + i - 1] : bn0;
      int mn  = meas_nums[b * Nn + i];
      int mn0 = meas_nums[b * Nn];
      sh_ch[b] = (i == 0) || (bn != bnp);
      int cb = bn - bn0; cb = cb < 0 ? 0 : (cb > NBt - 1 ? NBt - 1 : cb);
      int cm = mn - mn0; cm = cm < 0 ? 0 : (cm > NMs - 1 ? NMs - 1 : cm);
      sh_cb[b] = cb; sh_cm[b] = cm; sh_pb[b] = bnp;
    }
    __syncthreads();
    if (tid == 0) { int a = 0; for (int b = 0; b < 8; ++b) a |= sh_ch[b]; sh_any = a; }
    __syncthreads();

    if (sh_any) {
      // -------- attention over previous beat --------
#pragma unroll 1
      for (int idx = tid; idx < 8 * Nn; idx += 256) {
        int b = idx >> 10, j = idx & (Nn - 1);
        bool msk = (j < i) && (beat_nums[b * Nn + j] == sh_pb[b]);
        float s = -1e10f;
        if (msk) {
          float xv[10]; float xs = 0.f;
#pragma unroll
          for (int o = 0; o < 10; ++o) { xv[o] = outw[(size_t)(b * Nn + j) * 11 + 1 + o]; xs += xv[o]; }
          if (xs != 0.f) {
            float acc = 0.f;
#pragma unroll 1
            for (int o = 0; o < 10; ++o) {
              float d = att_b[o];
#pragma unroll
              for (int p = 0; p < 10; ++p) d += att_W[o * 10 + p] * xv[p];
              acc += tanhf(d) * att_ctx[o];
            }
            s = acc;
          }
        }
        simw[b * Nn + j] = s;
      }
      __syncthreads();
      { // per-batch max + sumexp; wave b <-> batch b
        int b = wave;
        float mx = -1e30f;
        for (int j = lane; j < Nn; j += 32) mx = fmaxf(mx, simw[b * Nn + j]);
        sh_red[b][lane] = mx;
        __syncthreads();
        if (lane == 0) { float mm = -1e30f;
          for (int l = 0; l < 32; ++l) mm = fmaxf(mm, sh_red[b][l]); sh_mx[b] = mm; }
        __syncthreads();
        float mm = sh_mx[b], se = 0.f;
        for (int j = lane; j < Nn; j += 32) se += __expf(simw[b * Nn + j] - mm);
        sh_red[b][lane] = se;
        __syncthreads();
        if (lane == 0) { float t = 0.f;
          for (int l = 0; l < 32; ++l) t += sh_red[b][l]; sh_se[b] = t; }
      }
      if (tid < 80) sh_res[tid / 10][tid % 10] = 0.f;
      __syncthreads();
      { // weighted sum
        int b = wave; float mm = sh_mx[b], inv = 1.f / sh_se[b];
        float r[10];
#pragma unroll
        for (int o = 0; o < 10; ++o) r[o] = 0.f;
#pragma unroll 1
        for (int j = lane; j < Nn; j += 32) {
          bool msk = (j < i) && (beat_nums[b * Nn + j] == sh_pb[b]);
          if (!msk) continue;
          float w = __expf(simw[b * Nn + j] - mm) * inv;
#pragma unroll
          for (int o = 0; o < 10; ++o) r[o] += w * outw[(size_t)(b * Nn + j) * 11 + 1 + o];
        }
#pragma unroll
        for (int o = 0; o < 10; ++o) atomicAdd(&sh_res[b][o], r[o]);
      }
      __syncthreads();
      if (tid < 80) { int b = tid / 10, o = tid % 10;
        if (sh_ch[b]) sh_cbr[b][o] = sh_res[b][o]; }
      __syncthreads();

      // -------- beat LSTM (biases + pz folded into Abe_bt) --------
#pragma unroll 1
      for (int idx = tid; idx < 8 * 512; idx += 256) {
        int b = idx >> 9, j = idx & 511;
        float v = Abe_bt[(size_t)(b * NBt + sh_cb[b]) * 512 + j]
                + Ame_bt[(size_t)(b * NMs + sh_cm[b]) * 512 + j];
        v += (float)wpbt16[j * 12 + 0] * sh_prev[b][0];
#pragma unroll
        for (int o = 0; o < 10; ++o) v += (float)wpbt16[j * 12 + 1 + o] * sh_cbr[b][o];
        sh_z[b][j] = v;
      }
      __syncthreads();
      { // th @ bt_Whh.T via WMMA
        v8f acc[4];
#pragma unroll
        for (int t = 0; t < 4; ++t) acc[t] = vzero8();
#pragma unroll
        for (int ks = 0; ks < 4; ++ks) {
          int k0 = ks * 32 + hi * 8;
          v16bf av = ldfrag(&sh_thbf[m][0], k0);
#pragma unroll
          for (int t = 0; t < 4; ++t) {
            const __bf16* wrow = wbthh16 + (size_t)((wave * 4 + t) * 16 + m) * 128;
            v16bf bv = ldfrag(wrow, k0);
            acc[t] = __builtin_amdgcn_wmma_f32_16x16x32_bf16(false, av, false, bv,
                                                             (short)0, acc[t], false, false);
          }
        }
        if (hi == 0) {
#pragma unroll
          for (int t = 0; t < 4; ++t)
#pragma unroll
            for (int r = 0; r < 8; ++r)
              sh_z[r][(wave * 4 + t) * 16 + m] += acc[t][r];
        }
      }
      if (tid < 8) sh_tempo[tid] = 0.f;
      __syncthreads();
#pragma unroll 1
      for (int idx = tid; idx < 8 * 128; idx += 256) {
        int b = idx >> 7, k = idx & 127;
        float c2 = sigmf(sh_z[b][128 + k]) * sh_tc[b][k]
                 + sigmf(sh_z[b][k]) * tanhf(sh_z[b][256 + k]);
        float h2 = sigmf(sh_z[b][384 + k]) * tanhf(c2);
        atomicAdd(&sh_tempo[b], h2 * btf_W[k]);
        if (sh_ch[b]) { sh_tc[b][k] = c2; sh_th[b][k] = h2; sh_thbf[b][k] = (__bf16)h2; }
      }
      __syncthreads();
      if (tid < 8 && sh_ch[tid]) sh_prev[tid][0] = sh_tempo[tid] + btf_b[0];
      __syncthreads();
    }

    // -------- output LSTM (biases + pz folded into Anote) --------
#pragma unroll 1
    for (int idx = tid; idx < 8 * 512; idx += 256) {
      int b = idx >> 9, j = idx & 511;
      float v = Anote[(size_t)(b * Nn + i) * 512 + j]
              + Abe_ol[(size_t)(b * NBt + sh_cb[b]) * 512 + j]
              + Ame_ol[(size_t)(b * NMs + sh_cm[b]) * 512 + j];
#pragma unroll
      for (int p = 0; p < 11; ++p) v += (float)Wp_ol[j][p] * sh_prev[b][p];
      sh_z[b][j] = v;
      if (i + 1 < Nn)   // overlap next step's row fetch with this step's work
        __builtin_prefetch(&Anote[(size_t)(b * Nn + i + 1) * 512 + j], 0, 1);
    }
    { // h @ ol_Whh.T with register-resident B fragments
      v8f acc[4];
#pragma unroll
      for (int t = 0; t < 4; ++t) acc[t] = vzero8();
#pragma unroll
      for (int ks = 0; ks < 4; ++ks) {
        v16bf av = ldfrag(&sh_hbf[m][0], ks * 32 + hi * 8);
#pragma unroll
        for (int t = 0; t < 4; ++t)
          acc[t] = __builtin_amdgcn_wmma_f32_16x16x32_bf16(false, av, false, Bol[t][ks],
                                                           (short)0, acc[t], false, false);
      }
      __syncthreads();   // gather writes to sh_z complete
      if (hi == 0) {
#pragma unroll
        for (int t = 0; t < 4; ++t)
#pragma unroll
          for (int r = 0; r < 8; ++r)
            sh_z[r][(wave * 4 + t) * 16 + m] += acc[t][r];
      }
    }
    __syncthreads();
#pragma unroll 1
    for (int idx = tid; idx < 8 * 128; idx += 256) {
      int b = idx >> 7, k = idx & 127;
      float c2 = sigmf(sh_z[b][128 + k]) * sh_c[b][k]
               + sigmf(sh_z[b][k]) * tanhf(sh_z[b][256 + k]);
      float h2 = sigmf(sh_z[b][384 + k]) * tanhf(c2);
      sh_c[b][k] = c2; sh_h[b][k] = h2; sh_hbf[b][k] = (__bf16)h2;
    }
    __syncthreads();
    if (tid < 80) {
      int b = tid / 10, o = tid % 10;
      float d = fc_b[o];
      for (int k = 0; k < 128; ++k) d += fc_W[o * 128 + k] * sh_h[b][k];
      sh_prev[b][1 + o] = d;
    }
    __syncthreads();
    if (tid < 88) { int b = tid / 11, d = tid % 11;
      outw[(size_t)(b * Nn + i) * 11 + d] = sh_prev[b][d]; }
    __syncthreads();
  }
}

// ---------------------------------------------------------------------------
// Final zero-mask on note rows + copy to d_out
// ---------------------------------------------------------------------------
__global__ __launch_bounds__(256)
void k_finalize(const float* __restrict__ note_emb, const float* __restrict__ ws,
                float* __restrict__ out) {
  int row = blockIdx.x * blockDim.x + threadIdx.x;
  if (row >= Bx * Nn) return;
  float s = 0.f;
  for (int k = 0; k < 256; ++k) s += note_emb[(size_t)row * 256 + k];
  const float* src = ws + WS_OUT + (size_t)row * 11;
#pragma unroll
  for (int d = 0; d < 11; ++d) out[(size_t)row * 11 + d] = (s == 0.f) ? 0.f : src[d];
}

// ---------------------------------------------------------------------------
static inline int cvt_blocks(int elems) {
  int b = (elems + 255) / 256;
  return b > 4096 ? 4096 : b;
}

extern "C" void kernel_launch(void* const* d_in, const int* in_sizes, int n_in,
                              void* d_out, int out_size, void* d_ws, size_t ws_size,
                              hipStream_t stream) {
  const float* note_emb    = (const float*)d_in[0];
  const float* beat_emb    = (const float*)d_in[1];
  const float* measure_emb = (const float*)d_in[2];
  const float* perf        = (const float*)d_in[3];
  const float* res_info    = (const float*)d_in[4];
  const int*   beat_nums   = (const int*)d_in[5];
  const int*   meas_nums   = (const int*)d_in[6];
  const float* style_W = (const float*)d_in[7];
  const float* style_b = (const float*)d_in[8];
  const float* md_Wih  = (const float*)d_in[9];
  const float* md_Whh  = (const float*)d_in[10];
  const float* md_bih  = (const float*)d_in[11];
  const float* md_bhh  = (const float*)d_in[12];
  const float* md_fcW  = (const float*)d_in[13];
  const float* md_fcb  = (const float*)d_in[14];
  const float* bt_Wih  = (const float*)d_in[15];
  const float* bt_Whh  = (const float*)d_in[16];
  const float* bt_bih  = (const float*)d_in[17];
  const float* bt_bhh  = (const float*)d_in[18];
  const float* ol_Wih  = (const float*)d_in[19];
  const float* ol_Whh  = (const float*)d_in[20];
  const float* ol_bih  = (const float*)d_in[21];
  const float* ol_bhh  = (const float*)d_in[22];
  const float* fc_W    = (const float*)d_in[23];
  const float* fc_b    = (const float*)d_in[24];
  const float* btf_W   = (const float*)d_in[25];
  const float* btf_b   = (const float*)d_in[26];
  const float* att_W   = (const float*)d_in[27];
  const float* att_b   = (const float*)d_in[28];
  const float* att_ctx = (const float*)d_in[29];
  float* ws  = (float*)d_ws;
  float* out = (float*)d_out;

  __bf16* note16   = (__bf16*)(ws + WS_NOTE16);
  __bf16* beat16   = (__bf16*)(ws + WS_BEAT16);
  __bf16* maug16   = (__bf16*)(ws + WS_MAUG16);
  __bf16* woln16   = (__bf16*)(ws + WS_WOLN16);
  __bf16* wolb16   = (__bf16*)(ws + WS_WOLB16);
  __bf16* wolm16   = (__bf16*)(ws + WS_WOLM16);
  __bf16* wbtb16   = (__bf16*)(ws + WS_WBTB16);
  __bf16* wbtm16   = (__bf16*)(ws + WS_WBTM16);
  __bf16* wolhh16  = (__bf16*)(ws + WS_WOLHH16);
  __bf16* wbthh16  = (__bf16*)(ws + WS_WBTHH16);
  __bf16* wpbt16   = (__bf16*)(ws + WS_WPBT16);

  k_pz<<<1, 256, 0, stream>>>(perf, style_W, style_b, ol_Wih, ol_bih, ol_bhh,
                              bt_Wih, bt_bih, bt_bhh, ws);

  // bf16 staging
  k_cvt<<<cvt_blocks(8192 * 256), 256, 0, stream>>>(note_emb, 256, 256, 256, 8192, note16);
  k_cvt<<<cvt_blocks(1024 * 256), 256, 0, stream>>>(beat_emb, 256, 256, 256, 1024, beat16);
  k_cvt<<<cvt_blocks(512 * 256), 256, 0, stream>>>(ol_Wih + 0,   OLW, 256, 256, 512, woln16);
  k_cvt<<<cvt_blocks(512 * 256), 256, 0, stream>>>(ol_Wih + 256, OLW, 256, 256, 512, wolb16);
  k_cvt<<<cvt_blocks(512 * 160), 256, 0, stream>>>(ol_Wih + 512, OLW, 130, 160, 512, wolm16);
  k_cvt<<<cvt_blocks(512 * 256), 256, 0, stream>>>(bt_Wih + 0,   BTW, 256, 256, 512, wbtb16);
  k_cvt<<<cvt_blocks(512 * 160), 256, 0, stream>>>(bt_Wih + 256, BTW, 130, 160, 512, wbtm16);
  k_cvt<<<cvt_blocks(512 * 128), 256, 0, stream>>>(ol_Whh, 128, 128, 128, 512, wolhh16);
  k_cvt<<<cvt_blocks(512 * 128), 256, 0, stream>>>(bt_Whh, 128, 128, 128, 512, wbthh16);
  k_cvt<<<cvt_blocks(512 * 12), 256, 0, stream>>>(bt_Wih + 386, BTW, 11, 12, 512, wpbt16);

  // precompute GEMMs; pz+bias rows folded into Anote / Abe_bt epilogues
  k_gemm16<<<512, 256, 0, stream>>>(note16, 256, woln16, 256, ws + WS_ANOTE, 8192, 8,
                                    ws + WS_APZ_OL, Nn);
  k_gemm16<<<64,  256, 0, stream>>>(beat16, 256, wolb16, 256, ws + WS_ABE_OL, 1024, 8,
                                    nullptr, 1);
  k_gemm16<<<64,  256, 0, stream>>>(beat16, 256, wbtb16, 256, ws + WS_ABE_BT, 1024, 8,
                                    ws + WS_APZ_BT, NBt);

  k_measure<<<1, 256, 0, stream>>>(measure_emb, res_info, md_Wih, md_Whh,
                                   md_bih, md_bhh, md_fcW, md_fcb, ws, out);
  k_cvt<<<cvt_blocks(256 * 160), 256, 0, stream>>>(ws + WS_MAUG, 130, 130, 160, 256, maug16);

  k_gemm16<<<16, 256, 0, stream>>>(maug16, 160, wolm16, 160, ws + WS_AME_OL, 256, 5,
                                   nullptr, 1);
  k_gemm16<<<16, 256, 0, stream>>>(maug16, 160, wbtm16, 160, ws + WS_AME_BT, 256, 5,
                                   nullptr, 1);

  k_mainloop<<<1, 256, 0, stream>>>(beat_nums, meas_nums, ol_Wih,
                                    fc_W, fc_b, btf_W, btf_b,
                                    att_W, att_b, att_ctx, ws);

  k_finalize<<<32, 256, 0, stream>>>(note_emb, ws, out);
}
